// LocalGlobalAttention_22746146799915
// MI455X (gfx1250) — matrix operand order repaired and checked by reading, not compiled, not run on hardware
//
#include <hip/hip_runtime.h>
#include <hip/hip_bf16.h>

// ---------------------------------------------------------------------------
// LocalGlobalAttention for MI455X (gfx1250, wave32, WMMA).
// Round 2: software-pipelined GEMMs (distinct B-frag registers so loads
// overlap WMMAs instead of s_wait_loadcnt 0 before each), and flash-attention
// K tiles staged through LDS with double-buffered GLOBAL_LOAD_ASYNC_TO_LDS
// (ASYNCcnt-tracked, s_wait_asynccnt pipelining).
// ---------------------------------------------------------------------------

typedef __attribute__((ext_vector_type(16))) __bf16 v16bf;
typedef __attribute__((ext_vector_type(8)))  float  v8f;

#define DEVI __device__ __forceinline__

constexpr int Bsz = 2;
constexpr int Lq  = 4096;
constexpr int HID = 2048;
constexpr int NH  = 32;
constexpr int HD  = 64;
constexpr int LOCAL_HEADS = 30;
constexpr int WIN = 1024;

// One WMMA operand fragment: 16 bf16 per lane = 8 VGPRs, loadable as 2x b128.
union FragBF { v16bf v; uint4 u[2]; };

DEVI __bf16 f2bf(float f) {
    union { float f; unsigned u; } in; in.f = f;
    unsigned r = in.u + 0x7FFFu + ((in.u >> 16) & 1u);   // round-to-nearest-even
    union { unsigned short s; __bf16 b; } out;
    out.s = (unsigned short)(r >> 16);
    return out.b;
}

DEVI v8f wmma_bf16(v16bf a, v16bf b, v8f c) {
    // D = A(16x32 bf16) * B(32x16 bf16) + C(16x16 f32)
    return __builtin_amdgcn_wmma_f32_16x16x32_bf16(
        /*neg_a=*/false, a, /*neg_b=*/false, b,
        /*c_mod=*/(short)0, c, /*reuse_a=*/false, /*reuse_b=*/false);
}

// Async-copy one contiguous 4KB tile (32 keys x 64 dims bf16) global -> LDS.
// Per ISA the LDS address is the VDST value plus the instruction offset, and the LDS-aperture
// flat address's low 32 bits are the wave-relative LDS offset, so the generic
// shared pointer's low word is the correct VDST payload. Tracked by ASYNCcnt.
DEVI void async_copy_4k(const __bf16* gsrc, __bf16* lds_dst, int lane) {
    unsigned ldsoff = (unsigned)(uintptr_t)lds_dst + (unsigned)(lane * 16);
    const char* g = (const char*)gsrc + lane * 16;
#pragma unroll
    for (int j = 0; j < 8; ++j) {
        asm volatile("global_load_async_to_lds_b128 %0, %1, off offset:%2"
                     :: "v"(ldsoff), "v"(g), "i"(j * 512) : "memory");
    }
}

// ---------------------------------------------------------------------------
__global__ void lga_cvt_f32_bf16(const float* __restrict__ s,
                                 __bf16* __restrict__ d, int n) {
    int i = blockIdx.x * blockDim.x + threadIdx.x;
    if (i < n) d[i] = f2bf(s[i]);
}

// ---------------------------------------------------------------------------
// QKV projection + RoPE.  One wave per (b, 16-row tile, head).
// A fragment (X rows): lane holds row = lane&15, K chunks {hi*8, 16+hi*8}.
// B fragment (W^T cols): lane holds col = lane&15 -> W row, K chunks {hi*8, 16+hi*8}.
// C layout: row M = r + 8*hi, col N = lane&15.
// 12 named B fragments per k-step keep 24 b128 loads in flight across the
// WMMA burst (no per-WMMA s_wait_loadcnt 0), and A(k+32) is preloaded.
// ---------------------------------------------------------------------------
__global__ __launch_bounds__(32) void lga_qkv_rope(
    const __bf16* __restrict__ X,
    const __bf16* __restrict__ Wq, const __bf16* __restrict__ Wk,
    const __bf16* __restrict__ Wv,
    __bf16* __restrict__ Qo, __bf16* __restrict__ Ko,
    __bf16* __restrict__ Vt)
{
    const int lane = threadIdx.x;
    const int qt = blockIdx.x, h = blockIdx.y, b = blockIdx.z;
    const int n = lane & 15, hi = lane >> 4;

    const __bf16* xrow = X + (size_t)(b * Lq + qt * 16 + n) * HID;
    const __bf16* wqh = Wq + (size_t)h * HD * HID;
    const __bf16* wkh = Wk + (size_t)h * HD * HID;
    const __bf16* wvh = Wv + (size_t)h * HD * HID;

    const v8f z8 = {0.f,0.f,0.f,0.f,0.f,0.f,0.f,0.f};
    v8f aq[4] = {z8,z8,z8,z8};
    v8f ak[4] = {z8,z8,z8,z8};
    v8f av[4] = {z8,z8,z8,z8};

    FragBF a;
    a.u[0] = *(const uint4*)(xrow + hi * 8);
    a.u[1] = *(const uint4*)(xrow + 16 + hi * 8);

    for (int k0 = 0; k0 < HID; k0 += 32) {
        FragBF w[12];
#pragma unroll
        for (int t = 0; t < 4; ++t) {
            const size_t wo = (size_t)(t * 16 + n) * HID + k0 + hi * 8;
            w[t].u[0]     = *(const uint4*)(wqh + wo);
            w[t].u[1]     = *(const uint4*)(wqh + wo + 16);
            w[4 + t].u[0] = *(const uint4*)(wkh + wo);
            w[4 + t].u[1] = *(const uint4*)(wkh + wo + 16);
            w[8 + t].u[0] = *(const uint4*)(wvh + wo);
            w[8 + t].u[1] = *(const uint4*)(wvh + wo + 16);
        }
        FragBF an;
        const bool more = (k0 + 32) < HID;
        if (more) {
            an.u[0] = *(const uint4*)(xrow + k0 + 32 + hi * 8);
            an.u[1] = *(const uint4*)(xrow + k0 + 48 + hi * 8);
            __builtin_prefetch(wqh + (size_t)n * HID + k0 + 64, 0, 1);
        }
#pragma unroll
        for (int t = 0; t < 4; ++t) {
            aq[t] = wmma_bf16(a.v, w[t].v,     aq[t]);
            ak[t] = wmma_bf16(a.v, w[4 + t].v, ak[t]);
            av[t] = wmma_bf16(a.v, w[8 + t].v, av[t]);
        }
        if (more) a = an;
    }

    // RoPE + store.  inv_freq[j] = exp(-j * ln(10000)/32), j = d & 31.
    const size_t qkbase = (size_t)(b * NH + h) * Lq;
    const size_t vtbase = (size_t)(b * NH + h) * HD;
#pragma unroll
    for (int t = 0; t < 4; ++t) {
        const int d = t * 16 + n;
        const int j = d & 31;
        const float invf = __expf(-(float)j * 0.28782313662f);
#pragma unroll
        for (int r = 0; r < 8; ++r) {
            const int pos = qt * 16 + r + hi * 8;
            const float th = (float)pos * invf;
            const float c = cosf(th), s = sinf(th);
            const float rq = (t < 2) ? -aq[t + 2][r] : aq[t - 2][r];
            const float rk = (t < 2) ? -ak[t + 2][r] : ak[t - 2][r];
            Qo[(qkbase + pos) * HD + d] = f2bf(aq[t][r] * c + rq * s);
            Ko[(qkbase + pos) * HD + d] = f2bf(ak[t][r] * c + rk * s);
            Vt[(vtbase + d) * Lq + pos] = f2bf(av[t][r]);   // V transposed [B,H,D,L]
        }
    }
}

// ---------------------------------------------------------------------------
// Flash attention, one wave per (b, head, 16-query tile).
// Local heads: ks = max(0, 1024*floor(qb/1024) - 512).  Global heads: ks = 0.
// Key blocks of 32: each 32x64 bf16 K tile is contiguous in [B,H,L,D], so it
// is staged into LDS with 8x global_load_async_to_lds_b128, double-buffered:
// issue copy for kb+32, s_wait_asynccnt 8 (async loads complete in order, so
// <=8 outstanding means the current tile has landed), compute from LDS.
// ---------------------------------------------------------------------------
__global__ __launch_bounds__(32) void lga_flash(
    const __bf16* __restrict__ Q, const __bf16* __restrict__ K,
    const __bf16* __restrict__ Vt, __bf16* __restrict__ O)
{
    const int lane = threadIdx.x;
    const int qt = blockIdx.x, h = blockIdx.y, b = blockIdx.z;
    const int n = lane & 15, hi = lane >> 4;
    const int hb8 = hi * 8, hb16 = hi * 16;
    const int qb = qt * 16, qe = qb + 16;

    int ks = 0;
    if (h < LOCAL_HEADS) {
        int bs = (qb / WIN) * WIN;
        ks = bs - WIN / 2;
        if (ks < 0) ks = 0;
    }

    const size_t hbase = (size_t)(b * NH + h);
    const __bf16* qrow = Q + (hbase * Lq + qb + n) * HD;
    FragBF a0, a1;                     // Q A-frags: dims 0..31 and 32..63
    a0.u[0] = *(const uint4*)(qrow + hb8);
    a0.u[1] = *(const uint4*)(qrow + 16 + hb8);
    a1.u[0] = *(const uint4*)(qrow + 32 + hb8);
    a1.u[1] = *(const uint4*)(qrow + 48 + hb8);

    const __bf16* kh = K  + hbase * Lq * HD;
    const __bf16* vh = Vt + hbase * HD * Lq;

    float mr[8], lr[8];
#pragma unroll
    for (int r = 0; r < 8; ++r) { mr[r] = -__builtin_inff(); lr[r] = 0.f; }
    const v8f z8 = {0.f,0.f,0.f,0.f,0.f,0.f,0.f,0.f};
    v8f acc[4] = {z8,z8,z8,z8};

    __shared__ __align__(16) __bf16 kls[2][32 * HD];   // double-buffered K tiles
    __shared__ __align__(16) __bf16 pls[16 * 32];      // P relayout buffer

    // Prologue: stage first K tile.
    async_copy_4k(kh + (size_t)ks * HD, &kls[0][0], lane);
    int cur = 0;

    for (int kb = ks; kb < qe; kb += 32) {
        const bool t1 = (kb + 16) < qe;             // second key tile in range?
        const bool has_next = (kb + 32) < qe;
        if (has_next)
            async_copy_4k(kh + (size_t)(kb + 32) * HD, &kls[cur ^ 1][0], lane);

        // Wait for the current tile; up to 8 newer async ops stay in flight.
        if (has_next) asm volatile("s_wait_asynccnt 0x8" ::: "memory");
        else          asm volatile("s_wait_asynccnt 0x0" ::: "memory");

        const __bf16* kt = &kls[cur][0];
        v8f s0 = z8, s1 = z8;
        {   // keys kb .. kb+15 (LDS rows 0..15)
            const __bf16* kr = kt + (size_t)n * HD + hb8;
            FragBF bk0, bk1;
            bk0.u[0] = *(const uint4*)(kr);      bk0.u[1] = *(const uint4*)(kr + 16);
            bk1.u[0] = *(const uint4*)(kr + 32); bk1.u[1] = *(const uint4*)(kr + 48);
            s0 = wmma_bf16(a0.v, bk0.v, s0);
            s0 = wmma_bf16(a1.v, bk1.v, s0);
        }
        if (t1) {   // keys kb+16 .. kb+31 (LDS rows 16..31)
            const __bf16* kr = kt + (size_t)(16 + n) * HD + hb8;
            FragBF bk0, bk1;
            bk0.u[0] = *(const uint4*)(kr);      bk0.u[1] = *(const uint4*)(kr + 16);
            bk1.u[0] = *(const uint4*)(kr + 32); bk1.u[1] = *(const uint4*)(kr + 48);
            s1 = wmma_bf16(a0.v, bk0.v, s1);
            s1 = wmma_bf16(a1.v, bk1.v, s1);
        }

        // Online softmax over this 32-key block (scale = 1/sqrt(64)).
#pragma unroll
        for (int r = 0; r < 8; ++r) {
            const int qi = qb + r + hb8;
            float v0 = s0[r] * 0.125f;
            if (qi < kb + n) v0 = -__builtin_inff();
            float v1 = -__builtin_inff();
            if (t1) {
                v1 = s1[r] * 0.125f;
                if (qi < kb + 16 + n) v1 = -__builtin_inff();
            }
            float mx = fmaxf(v0, v1);
#pragma unroll
            for (int o = 1; o < 16; o <<= 1) mx = fmaxf(mx, __shfl_xor(mx, o));
            const float mn = fmaxf(mr[r], mx);
            const float corr = __expf(mr[r] - mn);
            mr[r] = mn;
            const float p0 = __expf(v0 - mn);
            const float p1 = t1 ? __expf(v1 - mn) : 0.f;
            float ps = p0 + p1;
#pragma unroll
            for (int o = 1; o < 16; o <<= 1) ps += __shfl_xor(ps, o);
            lr[r] = lr[r] * corr + ps;
#pragma unroll
            for (int t = 0; t < 4; ++t) acc[t][r] = acc[t][r] * corr;
            const int m = r + hb8;
            pls[m * 32 + n]      = f2bf(p0);
            pls[m * 32 + 16 + n] = f2bf(p1);
        }
        __syncthreads();

        // Reload P in the A-matrix fragment layout (row = lane&15).
        FragBF ap;
        ap.u[0] = *(const uint4*)(&pls[n * 32 + hb8]);
        ap.u[1] = *(const uint4*)(&pls[n * 32 + 16 + hb8]);

        // O += P * V ; B frags per dim tile from transposed V (contiguous keys).
        FragBF bv[4];
#pragma unroll
        for (int dt = 0; dt < 4; ++dt) {
            const __bf16* vr = vh + (size_t)(dt * 16 + n) * Lq + kb + hb8;
            bv[dt].u[0] = *(const uint4*)(vr);
            bv[dt].u[1] = *(const uint4*)(vr + 16);
        }
#pragma unroll
        for (int dt = 0; dt < 4; ++dt)
            acc[dt] = wmma_bf16(ap.v, bv[dt].v, acc[dt]);
        __syncthreads();
        cur ^= 1;
    }

    // Normalize and store O in [B, L, H*D] layout for the output GEMM.
#pragma unroll
    for (int r = 0; r < 8; ++r) {
        const float inv = 1.f / lr[r];
        const size_t orow = ((size_t)b * Lq + qb + r + hb8) * (size_t)(NH * HD)
                            + (size_t)h * HD;
#pragma unroll
        for (int t = 0; t < 4; ++t)
            O[orow + t * 16 + n] = f2bf(acc[t][r] * inv);
    }
}

// ---------------------------------------------------------------------------
// Output projection: out = O * Wo^T + bo (f32 out).  16x64 strip per wave,
// same software pipelining as the QKV GEMM.
// ---------------------------------------------------------------------------
__global__ __launch_bounds__(32) void lga_out_proj(
    const __bf16* __restrict__ X, const __bf16* __restrict__ Wo,
    const float* __restrict__ bo, float* __restrict__ out)
{
    const int lane = threadIdx.x;
    const int qt = blockIdx.x, nb = blockIdx.y, b = blockIdx.z;
    const int n = lane & 15, hi = lane >> 4;

    const __bf16* xrow = X + (size_t)(b * Lq + qt * 16 + n) * HID;
    const __bf16* wb = Wo + (size_t)(nb * 64) * HID;

    const v8f z8 = {0.f,0.f,0.f,0.f,0.f,0.f,0.f,0.f};
    v8f acc[4] = {z8,z8,z8,z8};

    FragBF a;
    a.u[0] = *(const uint4*)(xrow + hi * 8);
    a.u[1] = *(const uint4*)(xrow + 16 + hi * 8);

    for (int k0 = 0; k0 < HID; k0 += 32) {
        FragBF w[4];
#pragma unroll
        for (int t = 0; t < 4; ++t) {
            const size_t wo = (size_t)(t * 16 + n) * HID + k0 + hi * 8;
            w[t].u[0] = *(const uint4*)(wb + wo);
            w[t].u[1] = *(const uint4*)(wb + wo + 16);
        }
        FragBF an;
        const bool more = (k0 + 32) < HID;
        if (more) {
            an.u[0] = *(const uint4*)(xrow + k0 + 32 + hi * 8);
            an.u[1] = *(const uint4*)(xrow + k0 + 48 + hi * 8);
            __builtin_prefetch(wb + (size_t)n * HID + k0 + 64, 0, 1);
        }
#pragma unroll
        for (int t = 0; t < 4; ++t)
            acc[t] = wmma_bf16(a.v, w[t].v, acc[t]);
        if (more) a = an;
    }

#pragma unroll
    for (int r = 0; r < 8; ++r) {
        const size_t row = (size_t)b * Lq + qt * 16 + r + hi * 8;
#pragma unroll
        for (int t = 0; t < 4; ++t) {
            const int col = nb * 64 + t * 16 + n;
            out[row * HID + col] = acc[t][r] + bo[col];
        }
    }
}

// ---------------------------------------------------------------------------
extern "C" void kernel_launch(void* const* d_in, const int* in_sizes, int n_in,
                              void* d_out, int out_size, void* d_ws, size_t ws_size,
                              hipStream_t stream)
{
    (void)in_sizes; (void)n_in; (void)out_size; (void)ws_size;

    const float* hs = (const float*)d_in[0];
    const float* Wq = (const float*)d_in[1];
    const float* Wk = (const float*)d_in[2];
    const float* Wv = (const float*)d_in[3];
    const float* Wo = (const float*)d_in[4];
    const float* bo = (const float*)d_in[5];
    float* out = (float*)d_out;

    char* ws = (char*)d_ws;
    size_t off = 0;
    auto alloc = [&](size_t bytes) -> char* {
        char* p = ws + off;
        off += (bytes + 255) & ~(size_t)255;
        return p;
    };
    __bf16* Xb  = (__bf16*)alloc((size_t)Bsz * Lq * HID * 2);
    __bf16* Wqb = (__bf16*)alloc((size_t)HID * HID * 2);
    __bf16* Wkb = (__bf16*)alloc((size_t)HID * HID * 2);
    __bf16* Wvb = (__bf16*)alloc((size_t)HID * HID * 2);
    __bf16* Wob = (__bf16*)alloc((size_t)HID * HID * 2);
    __bf16* Qb  = (__bf16*)alloc((size_t)Bsz * NH * Lq * HD * 2);
    __bf16* Kb  = (__bf16*)alloc((size_t)Bsz * NH * Lq * HD * 2);
    __bf16* Vtb = (__bf16*)alloc((size_t)Bsz * NH * HD * Lq * 2);
    __bf16* Ob  = (__bf16*)alloc((size_t)Bsz * Lq * HID * 2);

    const int nX = Bsz * Lq * HID;
    const int nW = HID * HID;
    lga_cvt_f32_bf16<<<(nX + 255) / 256, 256, 0, stream>>>(hs, Xb, nX);
    lga_cvt_f32_bf16<<<(nW + 255) / 256, 256, 0, stream>>>(Wq, Wqb, nW);
    lga_cvt_f32_bf16<<<(nW + 255) / 256, 256, 0, stream>>>(Wk, Wkb, nW);
    lga_cvt_f32_bf16<<<(nW + 255) / 256, 256, 0, stream>>>(Wv, Wvb, nW);
    lga_cvt_f32_bf16<<<(nW + 255) / 256, 256, 0, stream>>>(Wo, Wob, nW);

    dim3 g(Lq / 16, NH, Bsz);
    lga_qkv_rope<<<g, 32, 0, stream>>>(Xb, Wqb, Wkb, Wvb, Qb, Kb, Vtb);
    lga_flash<<<g, 32, 0, stream>>>(Qb, Kb, Vtb, Ob);

    dim3 go(Lq / 16, HID / 64, Bsz);
    lga_out_proj<<<go, 32, 0, stream>>>(Ob, Wob, bo, out);
}
